// PipelinedMoEBlock_4200478016071
// MI455X (gfx1250) — matrix operand order repaired and checked by reading, not compiled, not run on hardware
//
#include <hip/hip_runtime.h>
#include <hip/hip_bf16.h>
#include <math.h>

typedef __bf16 bf16;
typedef __attribute__((ext_vector_type(16))) __bf16 v16bf;
typedef __attribute__((ext_vector_type(8)))  float  v8f;

constexpr int Bc = 4, Sc = 1024, Dc = 1024, Hc = 16, HDc = 64, Ec = 8, Fc = 4096;
constexpr int NT = Bc * Sc;            // 4096 tokens

// ---------------------------------------------------------------- helpers
__device__ inline v8f wmma_bf16(v16bf a, v16bf b, v8f c) {
  return __builtin_amdgcn_wmma_f32_16x16x32_bf16(false, a, false, b, (short)0, c, false, false);
}

union FragU { v16bf v; uint4 u[2]; };

// A fragment (16x32 bf16, row-major, stride ldk) per CDNA5 layout.
__device__ inline v16bf load_a_frag(const bf16* base, int ldk, int lane) {
  int m = lane & 15, kh = lane >> 4;
  FragU r;
  const bf16* p = base + (size_t)m * ldk;
  r.u[0] = *(const uint4*)(p + kh * 8);
  r.u[1] = *(const uint4*)(p + 16 + kh * 8);
  return r.v;
}

// B fragment (32x16 bf16), element (k,n) at base[k*ldn + n].
__device__ inline v16bf load_b_frag(const bf16* base, int ldn, int lane) {
  int n = lane & 15, kh = lane >> 4;
  v16bf r;
#pragma unroll
  for (int j = 0; j < 16; ++j) r[j] = base[(size_t)(kh * 8 + (j & 7) + (j >> 3) * 16) * ldn + n];
  return r;
}

__device__ inline float gelu_tanh(float x) {
  float x3 = x * x * x;
  return 0.5f * x * (1.f + tanhf(0.7978845608028654f * (x + 0.044715f * x3)));
}

// async 16-byte global -> LDS copy, tracked by ASYNCcnt (CDNA5)
__device__ inline void async_g2l_b128(uint32_t lds_off, const void* gptr) {
  asm volatile("global_load_async_to_lds_b128 %0, %1, off"
               :: "v"(lds_off), "v"((uint64_t)(uintptr_t)gptr) : "memory");
}

// ---------------------------------------------------------------- fp32 -> bf16
__global__ __launch_bounds__(256) void cvt_kernel(const float* __restrict__ in,
                                                  bf16* __restrict__ out, size_t n) {
  size_t i = (size_t)blockIdx.x * 256 + threadIdx.x;
  size_t st = (size_t)gridDim.x * 256;
  for (; i < n; i += st) out[i] = (bf16)in[i];
}

// ---------------------------------------------------------------- LayerNorm -> bf16
__global__ __launch_bounds__(256) void ln_kernel(const float* __restrict__ x,
                                                 const float* __restrict__ g,
                                                 const float* __restrict__ bta,
                                                 bf16* __restrict__ out) {
  int row = blockIdx.x, tid = threadIdx.x;
  __shared__ float red[16];
  const float* xp = x + (size_t)row * Dc;
  float s = 0.f, s2 = 0.f;
  for (int i = tid; i < Dc; i += 256) { float v = xp[i]; s += v; s2 += v * v; }
  for (int o = 16; o; o >>= 1) { s += __shfl_xor(s, o, 32); s2 += __shfl_xor(s2, o, 32); }
  int lane = tid & 31, w = tid >> 5;
  if (lane == 0) { red[w] = s; red[8 + w] = s2; }
  __syncthreads();
  s = 0.f; s2 = 0.f;
#pragma unroll
  for (int i = 0; i < 8; ++i) { s += red[i]; s2 += red[8 + i]; }
  float mu = s / Dc;
  float inv = rsqrtf(s2 / Dc - mu * mu + 1e-5f);
  for (int i = tid; i < Dc; i += 256)
    out[(size_t)row * Dc + i] = (bf16)((xp[i] - mu) * inv * g[i] + bta[i]);
}

// ---------------------------------------------------------------- tiled WMMA GEMM
// C[M,N] = A[M,K] * B[K,N], block tile 128x128, 8 waves (4x2), wave tile 32x64.
// Staging: GLOBAL_LOAD_ASYNC_TO_LDS_B128, double-buffered, counted s_wait_asynccnt.
// EPI: 0 = store bf16, 1 = store f32 + residual, 2 = gelu->bf16, 3 = outF += gate*acc
template <int EPI>
__global__ __launch_bounds__(256) void gemm128(const bf16* __restrict__ A,
                                               const bf16* __restrict__ Bm,
                                               int K, int lda, int ldb,
                                               float* __restrict__ outF,
                                               bf16* __restrict__ outB, int ldo,
                                               const float* __restrict__ resid,
                                               const float* __restrict__ gates,
                                               int expert) {
  __shared__ alignas(16) bf16 As[2][128 * 32];
  __shared__ alignas(16) bf16 Bs[2][32 * 128];
  int tid = threadIdx.x, lane = tid & 31, wave = tid >> 5;
  int wm = wave & 3, wn = wave >> 2;
  int m0 = blockIdx.y * 128, n0 = blockIdx.x * 128;

  int ar = tid >> 1, ac = (tid & 1) * 16;   // A: 128 rows x 32 cols, 32B/thread
  int br = tid >> 3, bc = (tid & 7) * 16;   // B: 32 rows x 128 cols, 32B/thread

  // issue one stage (4 async b128 ops per thread)
  auto issue_stage = [&](int k0, int buf) {
    const bf16* pa = A + (size_t)(m0 + ar) * lda + k0 + ac;
    const bf16* pb = Bm + (size_t)(k0 + br) * ldb + n0 + bc;
    uint32_t la = (uint32_t)(uintptr_t)&As[buf][ar * 32 + ac];
    uint32_t lb = (uint32_t)(uintptr_t)&Bs[buf][br * 128 + bc];
    async_g2l_b128(la,      pa);
    async_g2l_b128(la + 16, pa + 8);
    async_g2l_b128(lb,      pb);
    async_g2l_b128(lb + 16, pb + 8);
  };

  v8f acc[2][4];
#pragma unroll
  for (int mi = 0; mi < 2; ++mi)
#pragma unroll
    for (int ni = 0; ni < 4; ++ni)
#pragma unroll
      for (int v = 0; v < 8; ++v) acc[mi][ni][v] = 0.f;

  int nk = K / 32;
  issue_stage(0, 0);
  for (int s = 0; s < nk; ++s) {
    if (s + 1 < nk) {
      issue_stage((s + 1) * 32, (s + 1) & 1);
      asm volatile("s_wait_asynccnt 0x4" ::: "memory");   // stage s landed (in-order)
    } else {
      asm volatile("s_wait_asynccnt 0x0" ::: "memory");
    }
    __syncthreads();
    int buf = s & 1;
    v16bf af[2], bfr[4];
    af[0] = load_a_frag(&As[buf][(wm * 32 + 0) * 32], 32, lane);
    af[1] = load_a_frag(&As[buf][(wm * 32 + 16) * 32], 32, lane);
#pragma unroll
    for (int ni = 0; ni < 4; ++ni)
      bfr[ni] = load_b_frag(&Bs[buf][wn * 64 + ni * 16], 128, lane);
#pragma unroll
    for (int mi = 0; mi < 2; ++mi)
#pragma unroll
      for (int ni = 0; ni < 4; ++ni)
        acc[mi][ni] = wmma_bf16(af[mi], bfr[ni], acc[mi][ni]);
    __syncthreads();   // all waves done with buf before it is overwritten next iter
  }

  int kh = lane >> 4, nn = lane & 15;
#pragma unroll
  for (int mi = 0; mi < 2; ++mi)
#pragma unroll
    for (int ni = 0; ni < 4; ++ni)
#pragma unroll
      for (int v = 0; v < 8; ++v) {
        int row = m0 + wm * 32 + mi * 16 + v + 8 * kh;
        int col = n0 + wn * 64 + ni * 16 + nn;
        float x = acc[mi][ni][v];
        size_t oi = (size_t)row * ldo + col;
        if (EPI == 0) outB[oi] = (bf16)x;
        else if (EPI == 1) outF[oi] = x + resid[oi];
        else if (EPI == 2) outB[oi] = (bf16)gelu_tanh(x);
        else { float gw = gates[row * Ec + expert]; outF[oi] += gw * x; }
      }
}

// ---------------------------------------------------------------- flash attention
// One wave = one 16-row Q tile of one head; 8 waves / block, 512 blocks total.
__global__ __launch_bounds__(256) void attn_kernel(const bf16* __restrict__ qkv,
                                                   bf16* __restrict__ out) {
  __shared__ alignas(16) bf16 plds[8][16 * 32];
  int lane = threadIdx.x & 31, wave = threadIdx.x >> 5;
  int tile = blockIdx.x * 8 + wave;
  int qt = tile & 63;              // q-tile within sequence (S/16 = 64)
  int h  = (tile >> 6) & (Hc - 1);
  int b  = tile >> 10;
  int q0g = b * Sc + qt * 16;      // global token row of this Q tile
  const int ld = 3 * Dc;
  int m = lane & 15, kh = lane >> 4;

  // Q fragments (A layout), 2 K-chunks of 32 dims
  v16bf qf[2];
  {
    const bf16* qp = qkv + (size_t)(q0g + m) * ld + h * HDc;
#pragma unroll
    for (int c = 0; c < 2; ++c) {
      FragU u;
      u.u[0] = *(const uint4*)(qp + c * 32 + kh * 8);
      u.u[1] = *(const uint4*)(qp + c * 32 + 16 + kh * 8);
      qf[c] = u.v;
    }
  }

  v8f accO[4];
  float mrow[8], lrow[8];
#pragma unroll
  for (int t = 0; t < 4; ++t)
#pragma unroll
    for (int v = 0; v < 8; ++v) accO[t][v] = 0.f;
#pragma unroll
  for (int v = 0; v < 8; ++v) { mrow[v] = -3.0e38f; lrow[v] = 0.f; }

  int qend = qt * 16 + 15;
  int nblk = (qend + 32) / 32;     // 32-key blocks covering [0, qend]
  for (int kb = 0; kb < nblk; ++kb) {
    int k0 = kb * 32;
    // ---- scores: S = Q * K^T  (two 16-key n-tiles)
    v8f sc[2];
#pragma unroll
    for (int nt = 0; nt < 2; ++nt) {
      const bf16* kp = qkv + (size_t)(b * Sc + k0 + nt * 16 + m) * ld + Dc + h * HDc;
      FragU kb0, kb1;
      kb0.u[0] = *(const uint4*)(kp + kh * 8);
      kb0.u[1] = *(const uint4*)(kp + 16 + kh * 8);
      kb1.u[0] = *(const uint4*)(kp + 32 + kh * 8);
      kb1.u[1] = *(const uint4*)(kp + 32 + 16 + kh * 8);
      v8f z;
#pragma unroll
      for (int v = 0; v < 8; ++v) z[v] = 0.f;
      z = wmma_bf16(qf[0], kb0.v, z);
      sc[nt] = wmma_bf16(qf[1], kb1.v, z);
    }
    // ---- scale + causal mask
#pragma unroll
    for (int v = 0; v < 8; ++v) {
      int qi = qt * 16 + v + 8 * kh;
#pragma unroll
      for (int nt = 0; nt < 2; ++nt) {
        int kj = k0 + nt * 16 + m;
        float s = sc[nt][v] * 0.125f;
        sc[nt][v] = (kj <= qi) ? s : -3.0e38f;
      }
    }
    // ---- online softmax (row reductions across 16-lane groups)
    float alpha[8], rs[8];
#pragma unroll
    for (int v = 0; v < 8; ++v) {
      float mx = fmaxf(sc[0][v], sc[1][v]);
      mx = fmaxf(mx, __shfl_xor(mx, 8, 16));
      mx = fmaxf(mx, __shfl_xor(mx, 4, 16));
      mx = fmaxf(mx, __shfl_xor(mx, 2, 16));
      mx = fmaxf(mx, __shfl_xor(mx, 1, 16));
      float nm = fmaxf(mrow[v], mx);
      alpha[v] = __expf(mrow[v] - nm);
      mrow[v] = nm;
      sc[0][v] = __expf(sc[0][v] - nm);
      sc[1][v] = __expf(sc[1][v] - nm);
      float r = sc[0][v] + sc[1][v];
      r += __shfl_xor(r, 8, 16);
      r += __shfl_xor(r, 4, 16);
      r += __shfl_xor(r, 2, 16);
      r += __shfl_xor(r, 1, 16);
      rs[v] = r;
      lrow[v] = lrow[v] * alpha[v] + rs[v];
#pragma unroll
      for (int t = 0; t < 4; ++t) accO[t][v] *= alpha[v];
    }
    // ---- P (C-layout) -> LDS row-major [16][32] -> A fragment
#pragma unroll
    for (int v = 0; v < 8; ++v) {
      plds[wave][(v + 8 * kh) * 32 + 0 * 16 + m] = (bf16)sc[0][v];
      plds[wave][(v + 8 * kh) * 32 + 1 * 16 + m] = (bf16)sc[1][v];
    }
    asm volatile("s_wait_dscnt 0" ::: "memory");
    FragU pf;
    {
      const bf16* pp = &plds[wave][m * 32];
      pf.u[0] = *(const uint4*)(pp + kh * 8);
      pf.u[1] = *(const uint4*)(pp + 16 + kh * 8);
    }
    asm volatile("s_wait_dscnt 0" ::: "memory");
    // ---- O += P * V   (4 dim-tiles of 16)
#pragma unroll
    for (int t = 0; t < 4; ++t) {
      v16bf vf;
#pragma unroll
      for (int j = 0; j < 16; ++j)
        vf[j] = qkv[(size_t)(b * Sc + k0 + kh * 8 + (j & 7) + (j >> 3) * 16) * ld + 2 * Dc + h * HDc + t * 16 + m];
      accO[t] = wmma_bf16(pf.v, vf, accO[t]);
    }
  }
  // ---- normalize + store bf16 [NT, D]
#pragma unroll
  for (int v = 0; v < 8; ++v) {
    float inv = 1.f / lrow[v];
#pragma unroll
    for (int t = 0; t < 4; ++t)
      out[(size_t)(q0g + v + 8 * kh) * Dc + h * HDc + t * 16 + m] = (bf16)(accO[t][v] * inv);
  }
}

// ---------------------------------------------------------------- gating (top-2 softmax)
__global__ __launch_bounds__(256) void gate_kernel(const bf16* __restrict__ h,
                                                   const float* __restrict__ Wg,
                                                   float* __restrict__ gates) {
  int n = blockIdx.x * 256 + threadIdx.x;
  if (n >= NT) return;
  float lg[Ec];
#pragma unroll
  for (int e = 0; e < Ec; ++e) lg[e] = 0.f;
  for (int d = 0; d < Dc; ++d) {
    float hv = (float)h[(size_t)n * Dc + d];
#pragma unroll
    for (int e = 0; e < Ec; ++e) lg[e] += hv * Wg[d * Ec + e];
  }
  int i0 = 0;
#pragma unroll
  for (int e = 1; e < Ec; ++e) if (lg[e] > lg[i0]) i0 = e;
  int i1 = (i0 == 0) ? 1 : 0;
#pragma unroll
  for (int e = 0; e < Ec; ++e) if (e != i0 && lg[e] > lg[i1]) i1 = e;
  float e0 = 1.f, e1 = __expf(lg[i1] - lg[i0]);
  float den = e0 + e1;
#pragma unroll
  for (int e = 0; e < Ec; ++e) gates[(size_t)n * Ec + e] = 0.f;
  gates[(size_t)n * Ec + i0] = e0 / den;
  gates[(size_t)n * Ec + i1] = e1 / den;
}

// ---------------------------------------------------------------- launch
extern "C" void kernel_launch(void* const* d_in, const int* in_sizes, int n_in,
                              void* d_out, int out_size, void* d_ws, size_t ws_size,
                              hipStream_t stream) {
  const float* x     = (const float*)d_in[0];
  const float* ln1g  = (const float*)d_in[1];
  const float* ln1b  = (const float*)d_in[2];
  const float* Wqkv  = (const float*)d_in[3];
  const float* Wo    = (const float*)d_in[4];
  const float* ln2g  = (const float*)d_in[5];
  const float* ln2b  = (const float*)d_in[6];
  const float* Wg    = (const float*)d_in[7];
  const float* W1    = (const float*)d_in[8];
  const float* W2    = (const float*)d_in[9];
  float* out = (float*)d_out;

  char* ws = (char*)d_ws;
  size_t o = 0;
  auto take = [&](size_t bytes) { char* p = ws + o; o += (bytes + 255) & ~(size_t)255; return p; };
  bf16*  wqkv_b = (bf16*)take((size_t)Dc * 3 * Dc * 2);
  bf16*  wo_b   = (bf16*)take((size_t)Dc * Dc * 2);
  bf16*  w1_b   = (bf16*)take((size_t)Ec * Dc * Fc * 2);
  bf16*  w2_b   = (bf16*)take((size_t)Ec * Fc * Dc * 2);
  bf16*  h1     = (bf16*)take((size_t)NT * Dc * 2);
  bf16*  qkv    = (bf16*)take((size_t)NT * 3 * Dc * 2);
  bf16*  ao     = (bf16*)take((size_t)NT * Dc * 2);
  float* xr     = (float*)take((size_t)NT * Dc * 4);
  bf16*  h2     = (bf16*)take((size_t)NT * Dc * 2);
  float* gates  = (float*)take((size_t)NT * Ec * 4);
  bf16*  act    = (bf16*)take((size_t)NT * Fc * 2);

  // 1) weights -> bf16
  cvt_kernel<<<4096, 256, 0, stream>>>(Wqkv, wqkv_b, (size_t)Dc * 3 * Dc);
  cvt_kernel<<<4096, 256, 0, stream>>>(Wo,   wo_b,   (size_t)Dc * Dc);
  cvt_kernel<<<4096, 256, 0, stream>>>(W1,   w1_b,   (size_t)Ec * Dc * Fc);
  cvt_kernel<<<4096, 256, 0, stream>>>(W2,   w2_b,   (size_t)Ec * Fc * Dc);

  // 2) LN1
  ln_kernel<<<NT, 256, 0, stream>>>(x, ln1g, ln1b, h1);

  // 3) QKV GEMM: [4096,1024] x [1024,3072] -> bf16
  gemm128<0><<<dim3(3 * Dc / 128, NT / 128), 256, 0, stream>>>(
      h1, wqkv_b, Dc, Dc, 3 * Dc, nullptr, qkv, 3 * Dc, nullptr, nullptr, 0);

  // 4) attention
  attn_kernel<<<NT * Hc / (16 * 8), 256, 0, stream>>>(qkv, ao);

  // 5) Wo GEMM + residual -> xr (f32)
  gemm128<1><<<dim3(Dc / 128, NT / 128), 256, 0, stream>>>(
      ao, wo_b, Dc, Dc, Dc, xr, nullptr, Dc, x, nullptr, 0);

  // 6) LN2
  ln_kernel<<<NT, 256, 0, stream>>>(xr, ln2g, ln2b, h2);

  // 7) gating
  gate_kernel<<<NT / 256, 256, 0, stream>>>(h2, Wg, gates);

  // 8) out = xr, then accumulate gated expert outputs
  hipMemcpyAsync(out, xr, (size_t)NT * Dc * 4, hipMemcpyDeviceToDevice, stream);

  for (int e = 0; e < Ec; ++e) {
    gemm128<2><<<dim3(Fc / 128, NT / 128), 256, 0, stream>>>(
        h2, w1_b + (size_t)e * Dc * Fc, Dc, Dc, Fc, nullptr, act, Fc, nullptr, nullptr, 0);
    gemm128<3><<<dim3(Dc / 128, NT / 128), 256, 0, stream>>>(
        act, w2_b + (size_t)e * Fc * Dc, Fc, Fc, Dc, out, nullptr, Dc, nullptr, gates, e);
  }
}
